// BridgeAttention_48490180772617
// MI455X (gfx1250) — matrix operand order repaired and checked by reading, not compiled, not run on hardware
//
#include <hip/hip_runtime.h>

// ---------------------------------------------------------------------------
// BridgeAttention for MI455X (gfx1250): fully fused per-(batch,node) attention
// B=8, P=Q=96, N=1024, C=H=256, D=128.  One 256-thread block (8 wave32) per
// (b,n).  All intermediates live in LDS; matmuls use v_wmma_f32_16x16x32_bf16.
//
// A prep kernel converts/transposes the four weight matrices into bf16
// B-fragment layout (Wt[h][k], column-major over K) in d_ws; the main kernel
// then builds every weight fragment with two contiguous global_load_b128s.
// ---------------------------------------------------------------------------

typedef __attribute__((ext_vector_type(16))) __bf16 v16bf;
typedef __attribute__((ext_vector_type(8)))  float  v8f;

static constexpr int Bc = 8;
static constexpr int Pp = 96;
static constexpr int Qq = 96;
static constexpr int Nn = 1024;
static constexpr int Cc = 256;   // HIDDEN (== H)
static constexpr int Dd = 128;   // STE

// ---- workspace layout (bf16 elements) --------------------------------------
static constexpr int WQT_OFF = 0;                 // WqT [256][128]
static constexpr int WKT_OFF = WQT_OFF + 256*128; // WkT [256][128]
static constexpr int WVT_OFF = WKT_OFF + 256*128; // WvT [256][256]
static constexpr int WOT_OFF = WVT_OFF + 256*256; // WoT [256][256]
static constexpr int WS_ELEMS = WOT_OFF + 256*256;       // 196608 elems = 384 KB

// ---- LDS layout (bytes) ----------------------------------------------------
static constexpr int SQ_OFF  = 0;                       // q [96][256] bf16 ; later: probs [96][96] bf16
static constexpr int SK_OFF  = SQ_OFF  + 96*256*2;      // k   [96][256] bf16
static constexpr int SVT_OFF = SK_OFF  + 96*256*2;      // v^T [256][96] bf16
static constexpr int SS_OFF  = SVT_OFF + 256*96*2;      // s   [96][96]  f32
static constexpr int SIN_OFF = SS_OFF  + 96*96*4;       // in  [96][256] bf16 (staging / attn-out)
static constexpr int LDS_BYTES = SIN_OFF + 96*256*2;    // = 233472 B (<= 320 KB/WGP)

__device__ __forceinline__ v8f wmma_bf(const v16bf& a, const v16bf& b, const v8f& c) {
  // D = A(16x32 bf16) * B(32x16 bf16) + C(16x16 f32)
  return __builtin_amdgcn_wmma_f32_16x16x32_bf16(
      /*neg_a=*/false, a, /*neg_b=*/false, b,
      /*c_mod=*/(short)0, c, /*reuse_a=*/false, /*reuse_b=*/false);
}

// A-fragment (16x32, bf16): lane L holds row M = row0 + L%16.
// Element e maps to K = k0 + (L/16)*16 + e  -> two contiguous
// 8-element runs per lane (-> 2x ds_load_b128).
__device__ __forceinline__ v16bf load_a_frag(const __bf16* base, int ld,
                                             int row0, int k0, int lane) {
  const int m = row0 + (lane & 15);
  const int half = lane >> 4;
  const __bf16* p0 = base + m * ld + k0 + half * 16;
  const __bf16* p1 = p0 + 8;
  v16bf a;
#pragma unroll
  for (int i = 0; i < 8; ++i) { a[i] = p0[i]; a[8 + i] = p1[i]; }
  return a;
}

// B-fragment (32x16, bf16) from a column-major (over K) buffer with K-stride
// ldK: lane L holds column N = col0 + L%16, element e -> K = k0 + (L/16)*16+e.
// 16 contiguous bf16 per lane -> 2x (ds|global)_load_b128.
__device__ __forceinline__ v16bf load_b_frag(const __bf16* base, int ldK,
                                             int col0, int k0, int lane) {
  const __bf16* p = base + (size_t)(col0 + (lane & 15)) * ldK + k0 + ((lane >> 4) << 4);
  v16bf b;
#pragma unroll
  for (int i = 0; i < 16; ++i) b[i] = p[i];
  return b;
}

// Stage a 96xCOLS f32 global slice (row stride rowStride elems) into LDS bf16.
// float4 global loads + packed 8B LDS stores.
template <int COLS>
__device__ __forceinline__ void stage_in(__bf16* dst, const float* __restrict__ src,
                                         size_t rowStride, int tid) {
  __syncthreads();  // previous readers of dst are done
  constexpr int TOT4 = 96 * COLS / 4;
#pragma unroll 4
  for (int i4 = tid; i4 < TOT4; i4 += 256) {
    const int e = i4 * 4;
    const int r = e / COLS;          // COLS is a power of two -> shift
    const int c = e & (COLS - 1);
    const float4 f = *reinterpret_cast<const float4*>(src + (size_t)r * rowStride + c);
    union { uint2 u; __bf16 h[4]; } pk;
    pk.h[0] = (__bf16)f.x; pk.h[1] = (__bf16)f.y;
    pk.h[2] = (__bf16)f.z; pk.h[3] = (__bf16)f.w;
    *reinterpret_cast<uint2*>(dst + e) = pk.u;
  }
  __syncthreads();  // staged data visible to all waves
}

// Projection GEMM:  sOut(96x256) = (sA(96xKDIM) @ W(KDIMx256) + bias) * scale.
// Wave w owns output column tiles 2w, 2w+1 (16 wide each), all 6 row tiles.
// No barriers: A from LDS, B (pre-transposed bf16 weights) from global (L2).
template <int KDIM, bool TRANSPOSED_OUT>
__device__ void proj_gemm(const __bf16* __restrict__ Wt, const float* __restrict__ bias,
                          float scale, const __bf16* sA, __bf16* sOut, int tid) {
  const int lane = tid & 31;
  const int w = tid >> 5;
  const int nt0 = w * 2;
  v8f acc[12];
#pragma unroll
  for (int i = 0; i < 12; ++i) acc[i] = v8f{};

#pragma unroll
  for (int kc = 0; kc < KDIM / 32; ++kc) {
    const v16bf b0 = load_b_frag(Wt, KDIM, nt0 * 16, kc * 32, lane);
    const v16bf b1 = load_b_frag(Wt, KDIM, nt0 * 16 + 16, kc * 32, lane);
#pragma unroll
    for (int mt = 0; mt < 6; ++mt) {
      const v16bf a = load_a_frag(sA, KDIM, mt * 16, kc * 32, lane);
      acc[mt * 2 + 0] = wmma_bf(a, b0, acc[mt * 2 + 0]);
      acc[mt * 2 + 1] = wmma_bf(a, b1, acc[mt * 2 + 1]);
    }
  }

  // Epilogue: C/D layout -> VGPR r, lane L: M = (L/16)*8 + r, N = L%16.
  const int half = lane >> 4;
  const int nc = lane & 15;
#pragma unroll
  for (int j = 0; j < 2; ++j) {
    const int h = (nt0 + j) * 16 + nc;
    const float bb = bias[h];
#pragma unroll
    for (int mt = 0; mt < 6; ++mt)
#pragma unroll
      for (int r = 0; r < 8; ++r) {
        const int m = mt * 16 + half * 8 + r;
        const float val = (acc[mt * 2 + j][r] + bb) * scale;
        if (TRANSPOSED_OUT) sOut[h * 96 + m] = (__bf16)val;   // v stored [H][P]
        else                sOut[m * 256 + h] = (__bf16)val;  // q/k stored [M][H]
      }
  }
}

// Final GEMM: out(96x256, global f32) = sA(96x256) @ Wo + bo
__device__ void out_gemm(const __bf16* __restrict__ WoT, const float* __restrict__ bias,
                         const __bf16* sA, float* __restrict__ outBase,
                         size_t outRowStride, int tid) {
  const int lane = tid & 31;
  const int w = tid >> 5;
  const int nt0 = w * 2;
  v8f acc[12];
#pragma unroll
  for (int i = 0; i < 12; ++i) acc[i] = v8f{};

#pragma unroll
  for (int kc = 0; kc < 256 / 32; ++kc) {
    const v16bf b0 = load_b_frag(WoT, 256, nt0 * 16, kc * 32, lane);
    const v16bf b1 = load_b_frag(WoT, 256, nt0 * 16 + 16, kc * 32, lane);
#pragma unroll
    for (int mt = 0; mt < 6; ++mt) {
      const v16bf a = load_a_frag(sA, 256, mt * 16, kc * 32, lane);
      acc[mt * 2 + 0] = wmma_bf(a, b0, acc[mt * 2 + 0]);
      acc[mt * 2 + 1] = wmma_bf(a, b1, acc[mt * 2 + 1]);
    }
  }

  const int half = lane >> 4;
  const int nc = lane & 15;
#pragma unroll
  for (int j = 0; j < 2; ++j) {
    const int h = (nt0 + j) * 16 + nc;
    const float bb = bias[h];
#pragma unroll
    for (int mt = 0; mt < 6; ++mt)
#pragma unroll
      for (int r = 0; r < 8; ++r) {
        const int m = mt * 16 + half * 8 + r;
        outBase[(size_t)m * outRowStride + h] = acc[mt * 2 + j][r] + bb;
      }
  }
}

// ---------------------------------------------------------------------------
// Prep kernel: Wt[h][k] = (bf16)W[k][h] for Wq/Wk (128x256) and Wv/Wo (256x256)
// Coalesced f32 reads over h; one-time 384 KB of bf16 into d_ws (L2-resident).
// ---------------------------------------------------------------------------
__global__ void __launch_bounds__(256)
BridgeAttention_prep_weights(const float* __restrict__ Wq, const float* __restrict__ Wk,
                             const float* __restrict__ Wv, const float* __restrict__ Wo,
                             __bf16* __restrict__ ws) {
  const int idx = blockIdx.x * 256 + threadIdx.x;   // 0 .. WS_ELEMS-1
  if (idx < 2 * 256 * 128) {                        // Wq | Wk  (K=128, H=256)
    const int which = idx >> 15;                    // 0: Wq, 1: Wk
    const int local = idx & (256 * 128 - 1);
    const int k = local >> 8;
    const int h = local & 255;
    const float* W = which ? Wk : Wq;
    __bf16* dst = ws + (which ? WKT_OFF : WQT_OFF);
    dst[h * 128 + k] = (__bf16)W[local];            // W[k*256+h]
  } else {                                          // Wv | Wo  (K=256, H=256)
    const int rem = idx - 2 * 256 * 128;
    const int which = rem >> 16;                    // 0: Wv, 1: Wo
    const int local = rem & (256 * 256 - 1);
    const int k = local >> 8;
    const int h = local & 255;
    const float* W = which ? Wo : Wv;
    __bf16* dst = ws + (which ? WOT_OFF : WVT_OFF);
    dst[h * 256 + k] = (__bf16)W[local];
  }
}

__global__ void __launch_bounds__(256)
BridgeAttention_48490180772617_kernel(const float* __restrict__ enc,
                                      const float* __restrict__ ste_enc,
                                      const float* __restrict__ ste_dec,
                                      const __bf16* __restrict__ wsT,
                                      const float* __restrict__ bq,
                                      const float* __restrict__ bk,
                                      const float* __restrict__ bv,
                                      const float* __restrict__ bo,
                                      float* __restrict__ out) {
  extern __shared__ char smem[];
  __bf16* sQ  = (__bf16*)(smem + SQ_OFF);    // q   [96][256]
  __bf16* sPb = (__bf16*)(smem + SQ_OFF);    // probs [96][96] bf16 (aliases dead sQ)
  __bf16* sK  = (__bf16*)(smem + SK_OFF);    // k   [96][256]
  __bf16* sVt = (__bf16*)(smem + SVT_OFF);   // v^T [256][96]
  float*  sS  = (float*)(smem + SS_OFF);     // scores [96][96] f32
  __bf16* sIn = (__bf16*)(smem + SIN_OFF);   // input staging / attn-out [96][256]

  const int tid = threadIdx.x;
  const int lane = tid & 31;
  const int w = tid >> 5;
  const int bid = blockIdx.x;
  const int b = bid >> 10;        // N = 1024
  const int n = bid & 1023;

  const size_t steStride = (size_t)Nn * Dd;  // row (p/q) stride for ste tensors
  const size_t encStride = (size_t)Nn * Cc;  // row stride for enc / out
  const float* decBase = ste_dec + ((size_t)b * Qq * Nn + n) * Dd;
  const float* senBase = ste_enc + ((size_t)b * Pp * Nn + n) * Dd;
  const float* encBase = enc     + ((size_t)b * Pp * Nn + n) * Cc;
  float*       outBase = out     + ((size_t)b * Qq * Nn + n) * Cc;

  // ---- 1) q = (ste_dec @ Wq + bq) / sqrt(C)  (scale folded into q) --------
  stage_in<Dd>(sIn, decBase, steStride, tid);
  proj_gemm<Dd, false>(wsT + WQT_OFF, bq, 0.0625f, sIn, sQ, tid);

  // ---- 2) k = ste_enc @ Wk + bk ------------------------------------------
  stage_in<Dd>(sIn, senBase, steStride, tid);
  proj_gemm<Dd, false>(wsT + WKT_OFF, bk, 1.0f, sIn, sK, tid);

  // ---- 3) v = enc @ Wv + bv  (stored transposed [H][P]) -------------------
  stage_in<Cc>(sIn, encBase, encStride, tid);
  proj_gemm<Cc, true>(wsT + WVT_OFF, bv, 1.0f, sIn, sVt, tid);

  __syncthreads();  // sQ/sK/sVt complete

  // ---- 4) scores = q @ k^T  (96x96, K = 256), already scaled --------------
  // 36 output tiles distributed round-robin across the 8 waves.
  for (int t = w; t < 36; t += 8) {
    const int mt = t / 6;
    const int nt = t % 6;
    v8f acc{};
#pragma unroll
    for (int kc = 0; kc < 8; ++kc) {
      const v16bf a = load_a_frag(sQ, 256, mt * 16, kc * 32, lane);
      const v16bf bb = load_b_frag(sK, 256, nt * 16, kc * 32, lane);  // k[p][h]: col=p, K=h
      acc = wmma_bf(a, bb, acc);
    }
    const int half = lane >> 4;
    const int nc = lane & 15;
#pragma unroll
    for (int r = 0; r < 8; ++r)
      sS[(mt * 16 + half * 8 + r) * 96 + nt * 16 + nc] = acc[r];
  }
  __syncthreads();  // scores done; sQ is now dead -> region reused for probs

  // ---- 5) softmax over P, one thread per row; emit bf16 probs -------------
  if (tid < 96) {
    float* row = sS + tid * 96;
    __bf16* prow = sPb + tid * 96;
    float mx = -3.4e38f;
    for (int p = 0; p < 96; ++p) mx = fmaxf(mx, row[p]);
    float sum = 0.f;
    for (int p = 0; p < 96; ++p) {
      const float e = __expf(row[p] - mx);
      row[p] = e;
      sum += e;
    }
    const float inv = 1.0f / sum;
    for (int p = 0; p < 96; ++p) prow[p] = (__bf16)(row[p] * inv);
  }
  __syncthreads();

  // ---- 6) attn_out = probs @ v  (96x256, K = 96) --> sIn (bf16) -----------
#pragma unroll
  for (int j = 0; j < 2; ++j) {
    const int nt = w * 2 + j;
#pragma unroll
    for (int mt = 0; mt < 6; ++mt) {
      v8f acc{};
#pragma unroll
      for (int kc = 0; kc < 3; ++kc) {
        const v16bf a = load_a_frag(sPb, 96, mt * 16, kc * 32, lane);
        const v16bf bb = load_b_frag(sVt, 96, nt * 16, kc * 32, lane);  // v^T[h][p]: col=h, K=p
        acc = wmma_bf(a, bb, acc);
      }
      const int half = lane >> 4;
      const int nc = lane & 15;
      const int h = nt * 16 + nc;
#pragma unroll
      for (int r = 0; r < 8; ++r) {
        const int m = mt * 16 + half * 8 + r;
        sIn[m * 256 + h] = (__bf16)acc[r];
      }
    }
  }
  __syncthreads();  // attn-out complete before the output projection reads it

  // ---- 7) out = attn_out @ Wo + bo  -> global f32 -------------------------
  out_gemm(wsT + WOT_OFF, bo, sIn, outBase, encStride, tid);
}

extern "C" void kernel_launch(void* const* d_in, const int* in_sizes, int n_in,
                              void* d_out, int out_size, void* d_ws, size_t ws_size,
                              hipStream_t stream) {
  (void)in_sizes; (void)n_in; (void)out_size; (void)ws_size;
  const float* enc     = (const float*)d_in[0];
  const float* ste_enc = (const float*)d_in[1];
  const float* ste_dec = (const float*)d_in[2];
  const float* Wq      = (const float*)d_in[3];
  const float* bq      = (const float*)d_in[4];
  const float* Wk      = (const float*)d_in[5];
  const float* bk      = (const float*)d_in[6];
  const float* Wv      = (const float*)d_in[7];
  const float* bv      = (const float*)d_in[8];
  const float* Wo      = (const float*)d_in[9];
  const float* bo      = (const float*)d_in[10];
  float* outp = (float*)d_out;
  __bf16* wsT = (__bf16*)d_ws;   // 384 KB of pre-transposed bf16 weights

  static_assert(LDS_BYTES <= 320 * 1024, "LDS budget exceeded");
  (void)hipFuncSetAttribute((const void*)BridgeAttention_48490180772617_kernel,
                            hipFuncAttributeMaxDynamicSharedMemorySize, LDS_BYTES);

  // Pass 1: convert + transpose weights into B-fragment-friendly bf16 layout.
  BridgeAttention_prep_weights<<<dim3(WS_ELEMS / 256), dim3(256), 0, stream>>>(
      Wq, Wk, Wv, Wo, wsT);

  // Pass 2: fused attention, one workgroup per (batch, node).
  dim3 grid(Bc * Nn);   // 8192 workgroups
  dim3 block(256);      // 8 wave32
  BridgeAttention_48490180772617_kernel<<<grid, block, LDS_BYTES, stream>>>(
      enc, ste_enc, ste_dec, wsT, bq, bk, bv, bo, outp);
}
